// Llama_72258529788207
// MI455X (gfx1250) — matrix operand order repaired and checked by reading, not compiled, not run on hardware
//
#include <hip/hip_runtime.h>

typedef __attribute__((ext_vector_type(16))) _Float16 v16h;
typedef __attribute__((ext_vector_type(8)))  float    v8f;

#define Dm     2048
#define HQc    16
#define HDc    128
#define HIDc   8192
#define Sc     2048
#define Bc     2
#define TOK    (Bc * Sc)            // 4096 tokens
#define LN10000 9.210340371976184f
#define INV_SQRT_HD 0.08838834764831845f
#define NEG_INF (-__builtin_inff())

// ---------------------------------------------------------------- embedding
__global__ void embed_kernel(const int* __restrict__ ids,
                             const float* __restrict__ embs,
                             float* __restrict__ x) {
  int t = blockIdx.x;
  const float* src = embs + (size_t)ids[t] * Dm;
  float* dst = x + (size_t)t * Dm;
  for (int i = threadIdx.x; i < Dm; i += blockDim.x) dst[i] = src[i];
}

// ---------------------------------------------------------------- rmsnorm
// out16 != null -> write f16 activations; out32 != null -> write f32 (final)
__global__ __launch_bounds__(256) void rmsnorm_kernel(
    const float* __restrict__ x, const float* __restrict__ scale,
    _Float16* __restrict__ out16, float* __restrict__ out32) {
  int t = blockIdx.x;
  const float* row = x + (size_t)t * Dm;
  __shared__ float red[256];
  float ss = 0.f;
  for (int i = threadIdx.x; i < Dm; i += 256) { float v = row[i]; ss += v * v; }
  red[threadIdx.x] = ss;
  __syncthreads();
  for (int s = 128; s > 0; s >>= 1) {
    if (threadIdx.x < s) red[threadIdx.x] += red[threadIdx.x + s];
    __syncthreads();
  }
  float rms = rsqrtf(red[0] * (1.0f / Dm) + 1e-5f);
  for (int i = threadIdx.x; i < Dm; i += 256) {
    float v = row[i] * rms * scale[i];
    if (out32) out32[(size_t)t * Dm + i] = v;
    else       out16[(size_t)t * Dm + i] = (_Float16)v;
  }
}

// ---------------------------------------------------------------- WMMA GEMM
union FragU { uint4 u[2]; v16h v; };

__device__ inline v16h ld_frag(const _Float16* p0, const _Float16* p1) {
  FragU f;
  f.u[0] = *reinterpret_cast<const uint4*>(p0);
  f.u[1] = *reinterpret_cast<const uint4*>(p1);
  return f.v;
}

// C[M,N](f32) = A[M,K](f16) @ Bw[K,N](f32->f16), optional += (residual).
// Block: 256 thr (8 waves), tile 128x128, K-step 32.
// Wave: 32 rows x 64 cols -> 2 A-frags x 4 B-frags = 8 WMMAs / K-step.
__global__ __launch_bounds__(256) void gemm_f16_kernel(
    const _Float16* __restrict__ A, const float* __restrict__ Bw,
    float* __restrict__ C, int M, int N, int K, int addC) {
  __shared__ _Float16 sA[128 * 40];   // [m][k], stride 40 halves (16B-aligned rows)
  __shared__ _Float16 sB[128 * 40];   // [n][k], stride 40 halves
  int t = threadIdx.x, wave = t >> 5, lane = t & 31;
  int nBlk = blockIdx.x * 128, mBlk = blockIdx.y * 128;
  int waveM = wave & 3;               // 4 row-groups of 32
  int waveN = wave >> 2;              // 2 col-groups of 64
  v8f zero8 = {0,0,0,0,0,0,0,0};
  v8f acc[2][4];
#pragma unroll
  for (int h = 0; h < 2; ++h)
#pragma unroll
    for (int i = 0; i < 4; ++i) acc[h][i] = zero8;

  for (int kb = 0; kb < K; kb += 32) {
    // prefetch next K-step tiles into cache while this step computes
    if (kb + 32 < K) {
      if (t < 32)  __builtin_prefetch(&Bw[(size_t)(kb + 32 + t) * N + nBlk], 0, 0);
      if (t < 128) __builtin_prefetch(&A[(size_t)(mBlk + t) * K + kb + 32], 0, 0);
    }
    __syncthreads();
    // cooperative A load: 128x32 halves as 512 16B chunks
#pragma unroll
    for (int j = 0; j < 2; ++j) {
      int e8 = t + 256 * j;
      int m = e8 >> 2, kk = (e8 & 3) << 3;
      *reinterpret_cast<uint4*>(&sA[m * 40 + kk]) =
          *reinterpret_cast<const uint4*>(&A[(size_t)(mBlk + m) * K + kb + kk]);
    }
    // cooperative B load + f32->f16 convert: 32x128, coalesced along N
#pragma unroll
    for (int i = 0; i < 16; ++i) {
      int e = t + 256 * i;
      int r = e >> 7, c = e & 127;
      sB[c * 40 + r] = (_Float16)Bw[(size_t)(kb + r) * N + nBlk + c];
    }
    __syncthreads();

    int kh = (lane >> 4) << 3;        // A: elems0-7 K=kh..kh+7, elems8-15 K=16+kh..
    int mw0 = (waveM << 5) + (lane & 15);
    v16h af0 = ld_frag(&sA[mw0 * 40 + kh], &sA[mw0 * 40 + 16 + kh]);
    int mw1 = mw0 + 16;
    v16h af1 = ld_frag(&sA[mw1 * 40 + kh], &sA[mw1 * 40 + 16 + kh]);
    int kboff = (lane >> 4) << 3;     // B: elems0-7 K=kboff..kboff+7, elems8-15 K=16+kboff..
#pragma unroll
    for (int ns = 0; ns < 4; ++ns) {
      const _Float16* bp = &sB[((waveN << 6) + (ns << 4) + (lane & 15)) * 40 + kboff];
      v16h bf = ld_frag(bp, bp + 16);
      acc[0][ns] = __builtin_amdgcn_wmma_f32_16x16x32_f16(
          false, af0, false, bf, (short)0, acc[0][ns], false, false);
      acc[1][ns] = __builtin_amdgcn_wmma_f32_16x16x32_f16(
          false, af1, false, bf, (short)0, acc[1][ns], false, false);
    }
  }
  // epilogue: C VGPR r -> M = r + 8*(lane>=16)
#pragma unroll
  for (int h = 0; h < 2; ++h) {
#pragma unroll
    for (int ns = 0; ns < 4; ++ns) {
      int col = nBlk + (waveN << 6) + (ns << 4) + (lane & 15);
      int rbase = mBlk + (waveM << 5) + (h << 4) + ((lane >> 4) << 3);
#pragma unroll
      for (int r = 0; r < 8; ++r) {
        size_t idx = (size_t)(rbase + r) * N + col;
        float val = acc[h][ns][r];
        C[idx] = addC ? (C[idx] + val) : val;
      }
    }
  }
}

// Fused uv GEMM + SwiGLU: H[M,HID](f16) = silu(A@Wuv[:, :HID]) * (A@Wuv[:, HID:])
__global__ __launch_bounds__(256) void gemm_swiglu_kernel(
    const _Float16* __restrict__ A, const float* __restrict__ Wuv,
    _Float16* __restrict__ H, int M, int K) {
  __shared__ _Float16 sA[128 * 40];
  __shared__ _Float16 sBu[64 * 40];
  __shared__ _Float16 sBg[64 * 40];
  int t = threadIdx.x, wave = t >> 5, lane = t & 31;
  int nBlk = blockIdx.x * 64, mBlk = blockIdx.y * 128;
  v8f zero8 = {0,0,0,0,0,0,0,0};
  v8f accU[4], accG[4];
#pragma unroll
  for (int i = 0; i < 4; ++i) { accU[i] = zero8; accG[i] = zero8; }

  for (int kb = 0; kb < K; kb += 32) {
    if (kb + 32 < K) {
      if (t < 32) {
        __builtin_prefetch(&Wuv[(size_t)(kb + 32 + t) * (2 * HIDc) + nBlk], 0, 0);
        __builtin_prefetch(&Wuv[(size_t)(kb + 32 + t) * (2 * HIDc) + HIDc + nBlk], 0, 0);
      }
      if (t < 128) __builtin_prefetch(&A[(size_t)(mBlk + t) * K + kb + 32], 0, 0);
    }
    __syncthreads();
#pragma unroll
    for (int j = 0; j < 2; ++j) {
      int e8 = t + 256 * j;
      int m = e8 >> 2, kk = (e8 & 3) << 3;
      *reinterpret_cast<uint4*>(&sA[m * 40 + kk]) =
          *reinterpret_cast<const uint4*>(&A[(size_t)(mBlk + m) * K + kb + kk]);
    }
#pragma unroll
    for (int i = 0; i < 16; ++i) {
      int e = t + 256 * i;
      int half = e >> 11, e2 = e & 2047;
      int r = e2 >> 6, c = e2 & 63;
      float wv = Wuv[(size_t)(kb + r) * (2 * HIDc) + half * HIDc + nBlk + c];
      (half ? sBg : sBu)[c * 40 + r] = (_Float16)wv;
    }
    __syncthreads();

    int mw = (wave << 4) + (lane & 15);
    int kh = (lane >> 4) << 3;
    v16h af = ld_frag(&sA[mw * 40 + kh], &sA[mw * 40 + 16 + kh]);
    int kboff = (lane >> 4) << 3;
#pragma unroll
    for (int ns = 0; ns < 4; ++ns) {
      const _Float16* bu = &sBu[((ns << 4) + (lane & 15)) * 40 + kboff];
      const _Float16* bg = &sBg[((ns << 4) + (lane & 15)) * 40 + kboff];
      v16h bfu = ld_frag(bu, bu + 16);
      v16h bfg = ld_frag(bg, bg + 16);
      accU[ns] = __builtin_amdgcn_wmma_f32_16x16x32_f16(
          false, af, false, bfu, (short)0, accU[ns], false, false);
      accG[ns] = __builtin_amdgcn_wmma_f32_16x16x32_f16(
          false, af, false, bfg, (short)0, accG[ns], false, false);
    }
  }
#pragma unroll
  for (int ns = 0; ns < 4; ++ns) {
    int col = nBlk + (ns << 4) + (lane & 15);
    int rbase = mBlk + (wave << 4) + ((lane >> 4) << 3);
#pragma unroll
    for (int r = 0; r < 8; ++r) {
      float u = accU[ns][r], g = accG[ns][r];
      float hv = (u / (1.f + __expf(-u))) * g;      // silu(u)*g
      H[(size_t)(rbase + r) * HIDc + col] = (_Float16)hv;
    }
  }
}

// ---------------------------------------------------------------- RoPE on q, +1 (in place)
__global__ void rope_q_kernel(float* __restrict__ q) {
  int th = blockIdx.x;              // token*HQ + h
  int token = th >> 4, h = th & 15;
  int d = threadIdx.x;              // 0..127
  int pos = token & (Sc - 1);
  float* base = q + (size_t)token * Dm + h * HDc;
  float a = base[d];
  float b = base[d ^ 64];
  __syncthreads();                  // all pair-reads before any writes
  int fi = d & 63;
  float inv = __expf(-(float)fi * (LN10000 / 64.0f));
  float fr = (float)pos * inv;
  float c = cosf(fr), s = sinf(fr);
  float out = (d < 64) ? (a * c - b * s) : (a * c + b * s);
  base[d] = out + 1.0f;
}

// ---------------------------------------------------------------- k' = sum_g rope(k_g)+HKV ; v' = sum_g v_g
__global__ void prep_kv_kernel(const float* __restrict__ k, const float* __restrict__ v,
                               float* __restrict__ kp, float* __restrict__ vp) {
  int tkn = blockIdx.x;
  int d = threadIdx.x;              // 0..127
  int pos = tkn & (Sc - 1);
  int fi = d & 63;
  float inv = __expf(-(float)fi * (LN10000 / 64.0f));
  float fr = (float)pos * inv;
  float c = cosf(fr), s = sinf(fr);
  float ks = 0.f, vs = 0.f;
#pragma unroll
  for (int g = 0; g < 4; ++g) {
    const float* kb = k + (size_t)tkn * 512 + g * 128;
    float a = kb[d], bb = kb[d ^ 64];
    ks += (d < 64) ? (a * c - bb * s) : (a * c + bb * s);
    vs += v[(size_t)tkn * 512 + g * 128 + d];
  }
  kp[(size_t)tkn * HDc + d] = ks + 4.0f;  // + HKV (from sum_g of +1)
  vp[(size_t)tkn * HDc + d] = vs;
}

// ---------------------------------------------------------------- flash attention (fp32, single eff. KV head)
// grid (S/4, HQ, B), 128 thr = 4 waves, 1 query row per wave.
__global__ __launch_bounds__(128) void attn_kernel(
    const float* __restrict__ q, const float* __restrict__ kp,
    const float* __restrict__ vp, _Float16* __restrict__ o16) {
  int b = blockIdx.z, h = blockIdx.y;
  int wave = threadIdx.x >> 5, lane = threadIdx.x & 31;
  int row = blockIdx.x * 4 + wave;
  __shared__ float qsh[4][HDc];
  const float* qrow = q + ((size_t)(b * Sc + row)) * Dm + h * HDc;
#pragma unroll
  for (int cc = 0; cc < 4; ++cc) qsh[wave][lane * 4 + cc] = qrow[lane * 4 + cc];
  __syncthreads();
  const float* Kb = kp + (size_t)b * Sc * HDc;
  const float* Vb = vp + (size_t)b * Sc * HDc;
  float m = NEG_INF, l = 0.f;
  float4 o = make_float4(0.f, 0.f, 0.f, 0.f);
  for (int kt = 0; kt <= row; kt += 32) {
    int j = kt + lane;
    float sc = NEG_INF;
    if (j <= row) {                                   // causal mask
      const float* kr = Kb + (size_t)j * HDc;
      float acc = 0.f;
#pragma unroll
      for (int d = 0; d < HDc; d += 4) {
        float4 kv = *reinterpret_cast<const float4*>(kr + d);
        acc += qsh[wave][d] * kv.x + qsh[wave][d + 1] * kv.y +
               qsh[wave][d + 2] * kv.z + qsh[wave][d + 3] * kv.w;
      }
      // scores==HD -> -inf, then (s-1)/sqrt(HD)
      sc = (acc == 128.0f) ? NEG_INF : (acc - 1.0f) * INV_SQRT_HD;
    }
    float sm = sc;
#pragma unroll
    for (int off = 16; off > 0; off >>= 1) sm = fmaxf(sm, __shfl_xor(sm, off, 32));
    float mnew = fmaxf(m, sm);
    float scale = (m == NEG_INF) ? 0.f : __expf(m - mnew);
    float p = (sc == NEG_INF) ? 0.f : __expf(sc - mnew);
    float ps = p;
#pragma unroll
    for (int off = 16; off > 0; off >>= 1) ps += __shfl_xor(ps, off, 32);
    l = l * scale + ps;
    o.x *= scale; o.y *= scale; o.z *= scale; o.w *= scale;
#pragma unroll 8
    for (int jj = 0; jj < 32; ++jj) {
      float pj = __shfl(p, jj, 32);                   // uniform broadcast
      if (pj != 0.f) {
        const float* vr = Vb + (size_t)(kt + jj) * HDc + lane * 4;
        float4 vv = *reinterpret_cast<const float4*>(vr);
        o.x += pj * vv.x; o.y += pj * vv.y; o.z += pj * vv.z; o.w += pj * vv.w;
      }
    }
    m = mnew;
  }
  float invl = (l > 0.f) ? (1.0f / l) : 0.f;          // nan_to_num
  _Float16* orow = o16 + ((size_t)(b * Sc + row)) * Dm + h * HDc + lane * 4;
  orow[0] = (_Float16)(o.x * invl);
  orow[1] = (_Float16)(o.y * invl);
  orow[2] = (_Float16)(o.z * invl);
  orow[3] = (_Float16)(o.w * invl);
}

// ---------------------------------------------------------------- driver
extern "C" void kernel_launch(void* const* d_in, const int* in_sizes, int n_in,
                              void* d_out, int out_size, void* d_ws, size_t ws_size,
                              hipStream_t stream) {
  const int*   ids        = (const int*)d_in[0];
  const float* embs       = (const float*)d_in[1];
  const float* Wq         = (const float*)d_in[2];
  const float* Wk         = (const float*)d_in[3];
  const float* Wv         = (const float*)d_in[4];
  const float* Wo         = (const float*)d_in[5];
  const float* Wuv        = (const float*)d_in[6];
  const float* Wout       = (const float*)d_in[7];
  const float* attn_scale = (const float*)d_in[8];
  const float* ffn_scale  = (const float*)d_in[9];
  const float* final_sc   = (const float*)d_in[10];
  float* out = (float*)d_out;

  // workspace layout (~189 MB)
  char* w = (char*)d_ws;
  float*    x    = (float*)w;     w += (size_t)TOK * Dm * 4;
  _Float16* xn16 = (_Float16*)w;  w += (size_t)TOK * Dm * 2;
  float*    q    = (float*)w;     w += (size_t)TOK * Dm * 4;
  float*    kbuf = (float*)w;     w += (size_t)TOK * 512 * 4;
  float*    vbuf = (float*)w;     w += (size_t)TOK * 512 * 4;
  float*    kp   = (float*)w;     w += (size_t)TOK * HDc * 4;
  float*    vp   = (float*)w;     w += (size_t)TOK * HDc * 4;
  _Float16* o16  = (_Float16*)w;  w += (size_t)TOK * Dm * 2;
  _Float16* h16  = (_Float16*)w;  w += (size_t)TOK * HIDc * 2;

  embed_kernel<<<TOK, 256, 0, stream>>>(ids, embs, x);

  for (int li = 0; li < 4; ++li) {
    const float* Wq_l   = Wq   + (size_t)li * Dm * (HQc * HDc);
    const float* Wk_l   = Wk   + (size_t)li * Dm * 512;
    const float* Wv_l   = Wv   + (size_t)li * Dm * 512;
    const float* Wo_l   = Wo   + (size_t)li * (HQc * HDc) * Dm;
    const float* Wuv_l  = Wuv  + (size_t)li * Dm * (2 * HIDc);
    const float* Wout_l = Wout + (size_t)li * HIDc * Dm;

    rmsnorm_kernel<<<TOK, 256, 0, stream>>>(x, attn_scale + li * Dm, xn16, nullptr);

    gemm_f16_kernel<<<dim3(Dm / 128, TOK / 128), 256, 0, stream>>>(
        xn16, Wq_l, q, TOK, Dm, Dm, 0);
    gemm_f16_kernel<<<dim3(512 / 128, TOK / 128), 256, 0, stream>>>(
        xn16, Wk_l, kbuf, TOK, 512, Dm, 0);
    gemm_f16_kernel<<<dim3(512 / 128, TOK / 128), 256, 0, stream>>>(
        xn16, Wv_l, vbuf, TOK, 512, Dm, 0);

    rope_q_kernel<<<TOK * HQc, 128, 0, stream>>>(q);
    prep_kv_kernel<<<TOK, 128, 0, stream>>>(kbuf, vbuf, kp, vp);

    attn_kernel<<<dim3(Sc / 4, HQc, Bc), 128, 0, stream>>>(q, kp, vp, o16);

    // x += o @ Wo
    gemm_f16_kernel<<<dim3(Dm / 128, TOK / 128), 256, 0, stream>>>(
        o16, Wo_l, x, TOK, Dm, Dm, 1);

    rmsnorm_kernel<<<TOK, 256, 0, stream>>>(x, ffn_scale + li * Dm, xn16, nullptr);

    gemm_swiglu_kernel<<<dim3(HIDc / 64, TOK / 128), 256, 0, stream>>>(
        xn16, Wuv_l, h16, TOK, Dm);

    // x += h @ Wout
    gemm_f16_kernel<<<dim3(Dm / 128, TOK / 128), 256, 0, stream>>>(
        h16, Wout_l, x, TOK, Dm, HIDc, 1);
  }

  rmsnorm_kernel<<<TOK, 256, 0, stream>>>(x, final_sc, nullptr, out);
}
